// ProteinTransformerStudent_82068235092224
// MI455X (gfx1250) — matrix operand order repaired and checked by reading, not compiled, not run on hardware
//
#include <hip/hip_runtime.h>

#define B_  8
#define S_  1024
#define D_  512
#define H_  8
#define L_  6
#define F_  2048
#define V_  33
#define HD_ 64
#define MS_ (B_*S_)   // 8192 rows

#ifndef __has_builtin
#define __has_builtin(x) 0
#endif

#if defined(__HIP_DEVICE_COMPILE__) && \
    __has_builtin(__builtin_amdgcn_tensor_load_to_lds) && \
    __has_builtin(__builtin_amdgcn_s_wait_tensorcnt)
#define USE_TDM 1
#else
#define USE_TDM 0
#endif

typedef __bf16 bf16;
typedef __attribute__((ext_vector_type(16))) __bf16 v16bf;
typedef __attribute__((ext_vector_type(8)))  __bf16 v8bf;
typedef __attribute__((ext_vector_type(8)))  float  v8f;
typedef unsigned int u32x4 __attribute__((ext_vector_type(4)));
typedef int i32x4 __attribute__((ext_vector_type(4)));
typedef int i32x8 __attribute__((ext_vector_type(8)));

#define AS3 __attribute__((address_space(3)))

// ---------------- bf16 helpers ----------------
__device__ __forceinline__ bf16 f2bf(float f) {
    unsigned u = __builtin_bit_cast(unsigned, f);
    u += 0x7fffu + ((u >> 16) & 1u);           // round-to-nearest-even
    unsigned short h = (unsigned short)(u >> 16);
    return __builtin_bit_cast(bf16, h);
}
__device__ __forceinline__ v16bf mk16(v8bf lo, v8bf hi) {
    v16bf r;
#pragma unroll
    for (int i = 0; i < 8; ++i) { r[i] = lo[i]; r[i + 8] = hi[i]; }
    return r;
}
__device__ __forceinline__ v8f wmma_bf16(v16bf a, v16bf b, v8f c) {
    return __builtin_amdgcn_wmma_f32_16x16x32_bf16(
        false, a, false, b, (short)0, c, false, false);
}
__device__ __forceinline__ float halfmax16(float v) {
#pragma unroll
    for (int i = 8; i >= 1; i >>= 1) v = fmaxf(v, __shfl_xor(v, i, 16));
    return v;
}
__device__ __forceinline__ float halfsum16(float v) {
#pragma unroll
    for (int i = 8; i >= 1; i >>= 1) v += __shfl_xor(v, i, 16);
    return v;
}

#if USE_TDM
// Issue a 2D TDM tile load: 128 rows x 64 bf16, global row stride = strideElems.
// LDS destination gets 4 DWORDs of pad every 32 DWORDs -> row pitch 72 bf16.
__device__ __forceinline__ void tdm_load_2d(const bf16* gsrc, unsigned ldsOff,
                                            long long strideElems) {
    unsigned long long ga = (unsigned long long)(const void*)gsrc;
    u32x4 g0;
    g0[0] = 1u;                                    // count=1, user mode
    g0[1] = ldsOff;                                // lds_addr (bytes)
    g0[2] = (unsigned)(ga & 0xFFFFFFFFull);        // global_addr[31:0]
    g0[3] = (unsigned)((ga >> 32) & 0x1FFFFFFull) | (2u << 30);  // [56:32] | type=2
    const unsigned TD0 = 0x40000000u, TD1 = 0x40000000u;  // huge tensor dims (no OOB)
    i32x8 g1;
    g1[0] = (int)((1u << 16)            // data_size = 2 bytes
                | (1u << 20)            // pad_enable
                | (4u << 22)            // pad_interval: 32 DWORDs
                | (3u << 25));          // pad_amount: 4 DWORDs
    g1[1] = (int)((TD0 & 0xFFFFu) << 16);                 // tensor_dim0[15:0]
    g1[2] = (int)((TD0 >> 16) | ((TD1 & 0xFFFFu) << 16)); // td0 hi | td1 lo
    g1[3] = (int)((TD1 >> 16) | (64u << 16));             // td1 hi | tile_dim0=64
    g1[4] = (int)(128u);                                  // tile_dim1=128, tile_dim2=0
    unsigned long long st = (unsigned long long)strideElems;
    g1[5] = (int)(st & 0xFFFFFFFFull);                    // tensor_dim0_stride lo
    g1[6] = (int)((st >> 32) & 0xFFFFull);                // stride hi | dim1_stride lo
    g1[7] = 0;
    i32x4 z4 = {0, 0, 0, 0};
#if defined(__clang_major__) && (__clang_major__ >= 23)
    i32x8 z8 = {0, 0, 0, 0, 0, 0, 0, 0};
    __builtin_amdgcn_tensor_load_to_lds(g0, g1, z4, z4, z8, 0);
#else
    __builtin_amdgcn_tensor_load_to_lds(g0, g1, z4, z4, 0);
#endif
}
__device__ __forceinline__ unsigned lds_off(const void* p) {
    return (unsigned)(unsigned long long)(AS3 const char*)p;
}
#endif

// ---------------- weight conversion (fp32 [L][K][N] -> bf16 transposed [L][N][K]) ----
__global__ void cvt_t_kernel(const float* __restrict__ src, bf16* __restrict__ dst,
                             int K, int N, int nl) {
    long long i = (long long)blockIdx.x * blockDim.x + threadIdx.x;
    long long tot = (long long)nl * K * N;
    if (i >= tot) return;
    long long per = (long long)K * N;
    int l = (int)(i / per);
    long long r = i - (long long)l * per;
    int k = (int)(r % K);
    int n = (int)(r / K);
    dst[(long long)l * per + (long long)n * K + k] =
        f2bf(src[(long long)l * per + (long long)k * N + n]);
}
// Wmlm [D][V=33] -> transposed+padded [128][D] bf16 (zero rows beyond V)
__global__ void cvt_t_pad_mlm_kernel(const float* __restrict__ src, bf16* __restrict__ dst) {
    int i = blockIdx.x * blockDim.x + threadIdx.x;
    if (i >= 128 * D_) return;
    int k = i & (D_ - 1), n = i / D_;
    dst[(size_t)n * D_ + k] = (n < V_) ? f2bf(src[(size_t)k * V_ + n]) : f2bf(0.f);
}

// ---------------- embedding + RoPE ----------------
__global__ void embed_rope_kernel(const int* __restrict__ tokens,
                                  const float* __restrict__ emb,
                                  float* __restrict__ xf, bf16* __restrict__ xb) {
    int i = blockIdx.x * blockDim.x + threadIdx.x;   // over B*S*(D/2)
    if (i >= MS_ * (D_ / 2)) return;
    int d2 = i % (D_ / 2);
    int bs = i / (D_ / 2);
    int s  = bs % S_;
    int tok = tokens[bs];
    float x1 = emb[(size_t)tok * D_ + 2 * d2];
    float x2 = emb[(size_t)tok * D_ + 2 * d2 + 1];
    float invf = powf(10000.0f, -((float)(2 * d2) / (float)D_));
    float ang = (float)s * invf;
    float sn = sinf(ang), cs = cosf(ang);
    float o1 = x1 * cs - x2 * sn;
    float o2 = x1 * sn + x2 * cs;
    size_t o = (size_t)bs * D_ + 2 * d2;
    xf[o] = o1; xf[o + 1] = o2;
    xb[o] = f2bf(o1); xb[o + 1] = f2bf(o2);
}

// ---------------- WMMA GEMM  Y = A[M,K] @ Wt[N,K]^T + bias ----------------
// MODE: 0 = f32 out, 1 = bf16 out, 2 = bf16 out transposed as Vt[b,h,hd,s], 3 = gelu + bf16
template <int MODE>
__global__ __launch_bounds__(256) void gemm_wmma_kernel(
    const bf16* __restrict__ A, const bf16* __restrict__ Wt,
    const float* __restrict__ bias,
    float* __restrict__ outf, bf16* __restrict__ outb,
    int K, int Nact, int ldo)
{
    // KC=64 tiles, double-buffered, TDM-padded row pitch 72 bf16 (144 B)
    __shared__ bf16 As[2][128][72];
    __shared__ bf16 Bs[2][128][72];

    const int tid  = threadIdx.x;
    const int lane = tid & 31;
    const int wv   = tid >> 5;        // 8 waves
    const int wm   = wv >> 1;         // 0..3 -> 32-row slab
    const int wn   = wv & 1;          // 0..1 -> 64-col slab
    const bool hi  = lane >= 16;
    const int l15  = lane & 15;
    const int Mbase = blockIdx.y * 128;
    const int Nbase = blockIdx.x * 128;
    const bf16* Abase = A  + (size_t)Mbase * K;
    const bf16* Wbase = Wt + (size_t)Nbase * K;

    v8f acc[2][4] = {};

#if USE_TDM
    if (wv == 0) {                       // pre-issue first tiles into buffer 0
        tdm_load_2d(Abase, lds_off(&As[0][0][0]), K);
        tdm_load_2d(Wbase, lds_off(&Bs[0][0][0]), K);
    }
#endif
    int it = 0;
    for (int k0 = 0; k0 < K; k0 += 64, ++it) {
        const int nb = it & 1;
#if USE_TDM
        if (wv == 0) {
            if ((k0 + 64) < K) {
                tdm_load_2d(Abase + (k0 + 64), lds_off(&As[nb ^ 1][0][0]), K);
                tdm_load_2d(Wbase + (k0 + 64), lds_off(&Bs[nb ^ 1][0][0]), K);
                __builtin_amdgcn_s_wait_tensorcnt(2);   // current tiles complete
            } else {
                __builtin_amdgcn_s_wait_tensorcnt(0);   // last tiles complete
            }
        }
        __syncthreads();
#else
        // synchronous staging: 4 x 16B per thread per tile
#pragma unroll
        for (int i = 0; i < 4; ++i) {
            int idx = tid + i * 256;           // 1024 chunks of 8 bf16
            int row = idx >> 3, c8 = (idx & 7) * 8;
            *(v8bf*)&As[nb][row][c8] = *(const v8bf*)&Abase[(size_t)row * K + k0 + c8];
            *(v8bf*)&Bs[nb][row][c8] = *(const v8bf*)&Wbase[(size_t)row * K + k0 + c8];
        }
        __syncthreads();
#endif
        const int kb   = hi ? 8 : 0;    // A-layout half select
        const int koff = hi ? 8 : 0;    // B-layout half select
#pragma unroll
        for (int kk = 0; kk < 64; kk += 32) {
            v16bf af[2], bf4[4];
#pragma unroll
            for (int tm = 0; tm < 2; ++tm) {
                int m = wm * 32 + tm * 16 + l15;
                af[tm] = mk16(*(v8bf*)&As[nb][m][kk + kb],
                              *(v8bf*)&As[nb][m][kk + kb + 16]);
            }
#pragma unroll
            for (int tn = 0; tn < 4; ++tn) {
                int n = wn * 64 + tn * 16 + l15;
                bf4[tn] = mk16(*(v8bf*)&Bs[nb][n][kk + koff],
                               *(v8bf*)&Bs[nb][n][kk + koff + 16]);
            }
#pragma unroll
            for (int tm = 0; tm < 2; ++tm)
#pragma unroll
                for (int tn = 0; tn < 4; ++tn)
                    acc[tm][tn] = wmma_bf16(af[tm], bf4[tn], acc[tm][tn]);
        }
        __syncthreads();
    }

    // store (C layout: vgpr g -> row g + 8*hi, col = lane&15)
#pragma unroll
    for (int tm = 0; tm < 2; ++tm) {
#pragma unroll
        for (int tn = 0; tn < 4; ++tn) {
            int col = Nbase + wn * 64 + tn * 16 + l15;
#pragma unroll
            for (int g = 0; g < 8; ++g) {
                int row = Mbase + wm * 32 + tm * 16 + g + (hi ? 8 : 0);
                if (col < Nact) {
                    float v = acc[tm][tn][g] + (bias ? bias[col] : 0.f);
                    if (MODE == 3) v = 0.5f * v * (1.0f + erff(v * 0.70710678118f));
                    if (MODE == 0) {
                        outf[(size_t)row * ldo + col] = v;
                    } else if (MODE == 2) {
                        int b = row >> 10, s = row & (S_ - 1);
                        int h = col >> 6, hd = col & 63;
                        outb[(((size_t)(b * H_ + h) * HD_) + hd) * S_ + s] = f2bf(v);
                    } else {
                        outb[(size_t)row * ldo + col] = f2bf(v);
                    }
                }
            }
        }
    }
}

// ---------------- flash attention (bf16 in, bf16 ctx out) ----------------
__global__ __launch_bounds__(256) void attn_kernel(
    const bf16* __restrict__ q, const bf16* __restrict__ k,
    const bf16* __restrict__ vt, const int* __restrict__ mask,
    bf16* __restrict__ ctx)
{
    __shared__ bf16 Plds[8][16][32];   // per-wave P staging (C layout -> A layout)

    const int tid  = threadIdx.x;
    const int lane = tid & 31;
    const int wv   = tid >> 5;
    const bool hi  = lane >= 16;
    const int l15  = lane & 15;

    const int blk  = blockIdx.x;            // B*H*(S/128) = 512
    const int bh   = blk >> 3;
    const int qblk = blk & 7;
    const int b    = bh >> 3;
    const int h    = bh & 7;
    const int qbase = qblk * 128 + wv * 16;
    const float scale = 0.125f;             // 1/sqrt(64)
    const int kb = hi ? 8 : 0;

    const bf16* qp = q + ((size_t)(b * S_ + qbase + l15)) * D_ + h * HD_;
    v16bf qa[2];
    qa[0] = mk16(*(const v8bf*)(qp + kb),      *(const v8bf*)(qp + kb + 16));
    qa[1] = mk16(*(const v8bf*)(qp + 32 + kb), *(const v8bf*)(qp + 32 + kb + 16));

    float mrow[8], lrow[8];
#pragma unroll
    for (int g = 0; g < 8; ++g) { mrow[g] = -1e30f; lrow[g] = 0.f; }
    v8f acc[4] = {};

    const bf16* vbase = vt + (size_t)(b * H_ + h) * HD_ * S_;
    const int* mbase  = mask + (size_t)b * S_ * S_;

    for (int kc = 0; kc < S_; kc += 32) {
        v8f sc[2] = {};
#pragma unroll
        for (int t = 0; t < 2; ++t) {
            int key = kc + t * 16 + l15;
            const bf16* kp = k + ((size_t)(b * S_ + key)) * D_ + h * HD_;
            int koff = hi ? 8 : 0;
#pragma unroll
            for (int c = 0; c < 2; ++c) {
                v16bf bb = mk16(*(const v8bf*)(kp + c * 32 + koff),
                                *(const v8bf*)(kp + c * 32 + koff + 16));
                sc[t] = wmma_bf16(qa[c], bb, sc[t]);
            }
        }
        float p[2][8];
#pragma unroll
        for (int g = 0; g < 8; ++g) {
            int row = qbase + g + (hi ? 8 : 0);
#pragma unroll
            for (int t = 0; t < 2; ++t) {
                int key = kc + t * 16 + l15;
                float v = sc[t][g] * scale;
                if (mbase[(size_t)row * S_ + key] == 0) v = -1e9f;
                p[t][g] = v;
            }
        }
#pragma unroll
        for (int g = 0; g < 8; ++g) {
            float rm   = fmaxf(halfmax16(p[0][g]), halfmax16(p[1][g]));
            float mnew = fmaxf(mrow[g], rm);
            float corr = __expf(mrow[g] - mnew);
            mrow[g] = mnew;
            p[0][g] = __expf(p[0][g] - mnew);
            p[1][g] = __expf(p[1][g] - mnew);
            lrow[g] = lrow[g] * corr + halfsum16(p[0][g]) + halfsum16(p[1][g]);
#pragma unroll
            for (int nt = 0; nt < 4; ++nt) acc[nt][g] *= corr;
        }
        __syncthreads();
#pragma unroll
        for (int g = 0; g < 8; ++g) {
            int row = g + (hi ? 8 : 0);
            Plds[wv][row][l15]      = f2bf(p[0][g]);
            Plds[wv][row][16 + l15] = f2bf(p[1][g]);
        }
        __syncthreads();
        v16bf pa = mk16(*(v8bf*)&Plds[wv][l15][kb], *(v8bf*)&Plds[wv][l15][kb + 16]);
        int koff = hi ? 8 : 0;
#pragma unroll
        for (int nt = 0; nt < 4; ++nt) {
            const bf16* vp = vbase + (size_t)(nt * 16 + l15) * S_ + kc + koff;
            v16bf vb = mk16(*(const v8bf*)vp, *(const v8bf*)(vp + 16));
            acc[nt] = wmma_bf16(pa, vb, acc[nt]);
        }
    }

#pragma unroll
    for (int g = 0; g < 8; ++g) {
        int row = qbase + g + (hi ? 8 : 0);
        float inv = 1.0f / fmaxf(lrow[g], 1e-20f);
#pragma unroll
        for (int nt = 0; nt < 4; ++nt) {
            ctx[((size_t)(b * S_ + row)) * D_ + h * HD_ + nt * 16 + l15] =
                f2bf(acc[nt][g] * inv);
        }
    }
}

// ---------------- residual add + LayerNorm (dual output) ----------------
__global__ __launch_bounds__(256) void addln_kernel(
    const float* __restrict__ x, const float* __restrict__ r,
    const float* __restrict__ g, const float* __restrict__ bta,
    float* __restrict__ yf, bf16* __restrict__ yb)
{
    const int row = blockIdx.x, t = threadIdx.x;
    const float* xp = x + (size_t)row * D_;
    float v0 = xp[t]       + (r ? r[(size_t)row * D_ + t]       : 0.f);
    float v1 = xp[t + 256] + (r ? r[(size_t)row * D_ + t + 256] : 0.f);
    float s = v0 + v1, s2 = v0 * v0 + v1 * v1;
#pragma unroll
    for (int i = 16; i >= 1; i >>= 1) { s += __shfl_xor(s, i, 32); s2 += __shfl_xor(s2, i, 32); }
    __shared__ float shs[8], shs2[8];
    int wv = t >> 5, lane = t & 31;
    if (lane == 0) { shs[wv] = s; shs2[wv] = s2; }
    __syncthreads();
    if (wv == 0) {
        float a  = (lane < 8) ? shs[lane]  : 0.f;
        float a2 = (lane < 8) ? shs2[lane] : 0.f;
#pragma unroll
        for (int i = 4; i >= 1; i >>= 1) { a += __shfl_xor(a, i, 32); a2 += __shfl_xor(a2, i, 32); }
        if (lane == 0) { shs[0] = a; shs2[0] = a2; }
    }
    __syncthreads();
    float mean = shs[0] * (1.0f / D_);
    float var  = shs2[0] * (1.0f / D_) - mean * mean;
    float rinv = rsqrtf(var + 1e-5f);
    float o0 = (v0 - mean) * rinv * g[t]       + bta[t];
    float o1 = (v1 - mean) * rinv * g[t + 256] + bta[t + 256];
    size_t o = (size_t)row * D_;
    yf[o + t] = o0;        yf[o + t + 256] = o1;
    yb[o + t] = f2bf(o0);  yb[o + t + 256] = f2bf(o1);
}

// ---------------- host orchestration ----------------
extern "C" void kernel_launch(void* const* d_in, const int* in_sizes, int n_in,
                              void* d_out, int out_size, void* d_ws, size_t ws_size,
                              hipStream_t stream) {
    const int*   tokens = (const int*)  d_in[0];
    const int*   mask   = (const int*)  d_in[1];
    const float* emb    = (const float*)d_in[2];
    const float* Wq = (const float*)d_in[3];  const float* bq = (const float*)d_in[4];
    const float* Wk = (const float*)d_in[5];  const float* bk = (const float*)d_in[6];
    const float* Wv = (const float*)d_in[7];  const float* bv = (const float*)d_in[8];
    const float* Wo = (const float*)d_in[9];  const float* bo = (const float*)d_in[10];
    const float* W1 = (const float*)d_in[11]; const float* b1 = (const float*)d_in[12];
    const float* W2 = (const float*)d_in[13]; const float* b2 = (const float*)d_in[14];
    const float* g1 = (const float*)d_in[15]; const float* beta1 = (const float*)d_in[16];
    const float* g2 = (const float*)d_in[17]; const float* beta2 = (const float*)d_in[18];
    const float* gf = (const float*)d_in[19]; const float* bfb = (const float*)d_in[20];
    const float* Wmlm = (const float*)d_in[21]; const float* bmlm = (const float*)d_in[22];
    float* out = (float*)d_out;

    char* ws = (char*)d_ws;
    size_t off = 0;
    auto alloc = [&](size_t bytes) -> void* {
        void* p = ws + off;
        off = (off + bytes + 255) & ~(size_t)255;
        return p;
    };
    bf16* wq_bf = (bf16*)alloc((size_t)L_ * D_ * D_ * 2);   // transposed [L][N][K]
    bf16* wk_bf = (bf16*)alloc((size_t)L_ * D_ * D_ * 2);
    bf16* wv_bf = (bf16*)alloc((size_t)L_ * D_ * D_ * 2);
    bf16* wo_bf = (bf16*)alloc((size_t)L_ * D_ * D_ * 2);
    bf16* w1_bf = (bf16*)alloc((size_t)L_ * D_ * F_ * 2);   // [L][F][D]
    bf16* w2_bf = (bf16*)alloc((size_t)L_ * F_ * D_ * 2);   // [L][D][F]
    bf16* wm_bf = (bf16*)alloc((size_t)128 * D_ * 2);       // [128][D]
    float* xf   = (float*)alloc((size_t)MS_ * D_ * 4);
    bf16*  xb   = (bf16*) alloc((size_t)MS_ * D_ * 2);
    float* x2f  = (float*)alloc((size_t)MS_ * D_ * 4);
    bf16*  x2b  = (bf16*) alloc((size_t)MS_ * D_ * 2);
    bf16*  qb   = (bf16*) alloc((size_t)MS_ * D_ * 2);
    bf16*  kbuf = (bf16*) alloc((size_t)MS_ * D_ * 2);
    bf16*  vtb  = (bf16*) alloc((size_t)MS_ * D_ * 2);
    bf16*  ctxb = (bf16*) alloc((size_t)MS_ * D_ * 2);
    bf16*  hb   = (bf16*) alloc((size_t)MS_ * F_ * 2);
    float* tmpf = (float*)alloc((size_t)MS_ * D_ * 4);
    (void)ws_size; (void)in_sizes; (void)n_in; (void)out_size;

    // 1) weights -> bf16, transposed to [N][K]
    auto cvtT = [&](const float* s, bf16* d, int K, int N, int nl) {
        long long tot = (long long)nl * K * N;
        cvt_t_kernel<<<dim3((unsigned)((tot + 255) / 256)), dim3(256), 0, stream>>>(s, d, K, N, nl);
    };
    cvtT(Wq, wq_bf, D_, D_, L_);
    cvtT(Wk, wk_bf, D_, D_, L_);
    cvtT(Wv, wv_bf, D_, D_, L_);
    cvtT(Wo, wo_bf, D_, D_, L_);
    cvtT(W1, w1_bf, D_, F_, L_);
    cvtT(W2, w2_bf, F_, D_, L_);
    cvt_t_pad_mlm_kernel<<<dim3((128 * D_ + 255) / 256), dim3(256), 0, stream>>>(Wmlm, wm_bf);

    // 2) embedding + RoPE
    embed_rope_kernel<<<dim3(MS_ * (D_ / 2) / 256), dim3(256), 0, stream>>>(tokens, emb, xf, xb);

    // 3) transformer layers
    dim3 blk(256);
    dim3 gDD(D_ / 128, MS_ / 128);   // N=512
    dim3 gDF(F_ / 128, MS_ / 128);   // N=2048
    for (int l = 0; l < L_; ++l) {
        const bf16* wql = wq_bf + (size_t)l * D_ * D_;
        const bf16* wkl = wk_bf + (size_t)l * D_ * D_;
        const bf16* wvl = wv_bf + (size_t)l * D_ * D_;
        const bf16* wol = wo_bf + (size_t)l * D_ * D_;
        const bf16* w1l = w1_bf + (size_t)l * D_ * F_;
        const bf16* w2l = w2_bf + (size_t)l * F_ * D_;
        const float* bql = bq + (size_t)l * D_;
        const float* bkl = bk + (size_t)l * D_;
        const float* bvl = bv + (size_t)l * D_;
        const float* bol = bo + (size_t)l * D_;
        const float* b1l = b1 + (size_t)l * F_;
        const float* b2l = b2 + (size_t)l * D_;

        gemm_wmma_kernel<1><<<gDD, blk, 0, stream>>>(xb, wql, bql, nullptr, qb,   D_, D_, D_);
        gemm_wmma_kernel<1><<<gDD, blk, 0, stream>>>(xb, wkl, bkl, nullptr, kbuf, D_, D_, D_);
        gemm_wmma_kernel<2><<<gDD, blk, 0, stream>>>(xb, wvl, bvl, nullptr, vtb,  D_, D_, D_);

        attn_kernel<<<dim3(B_ * H_ * (S_ / 128)), blk, 0, stream>>>(qb, kbuf, vtb, mask, ctxb);

        gemm_wmma_kernel<0><<<gDD, blk, 0, stream>>>(ctxb, wol, bol, tmpf, nullptr, D_, D_, D_);
        addln_kernel<<<dim3(MS_), blk, 0, stream>>>(xf, tmpf, g1 + (size_t)l * D_,
                                                    beta1 + (size_t)l * D_, x2f, x2b);

        gemm_wmma_kernel<3><<<gDF, blk, 0, stream>>>(x2b, w1l, b1l, nullptr, hb, D_, F_, F_);
        gemm_wmma_kernel<0><<<gDD, blk, 0, stream>>>(hb, w2l, b2l, tmpf, nullptr, F_, D_, D_);
        addln_kernel<<<dim3(MS_), blk, 0, stream>>>(x2f, tmpf, g2 + (size_t)l * D_,
                                                    beta2 + (size_t)l * D_, xf, xb);
    }

    // 4) final LayerNorm (no residual)
    addln_kernel<<<dim3(MS_), blk, 0, stream>>>(xf, nullptr, gf, bfb, x2f, x2b);

    // 5) MLM head: [MS,512] @ Wt[128(pad of 33)][512] -> out [MS,33]
    gemm_wmma_kernel<0><<<dim3(1, MS_ / 128), blk, 0, stream>>>(
        x2b, wm_bf, bmlm, out, nullptr, D_, V_, V_);
}
